// GATConvSingle_61435212202078
// MI455X (gfx1250) — matrix operand order repaired and checked by reading, not compiled, not run on hardware
//
#include <hip/hip_runtime.h>
#include <hip/hip_bf16.h>

typedef __attribute__((ext_vector_type(16))) _Float16 v16h;
typedef __attribute__((ext_vector_type(8)))  float    v8f;

#define FDIM 256
#define CDIM 128
#define LEAKY 0.2f
// B-pack geometry: 8 n-tiles x 8 k-blocks x 32 lanes x 16 halves
#define BPACK_HALVES (8 * 8 * 32 * 16)

// ---------------------------------------------------------------------------
// Kernel 0: swizzle W_v (256x128 f32) into per-lane WMMA B packing (f16).
// B tile (32x16): lanes 0-15 hold K=kk+0..15, lanes 16-31 hold K=kk+16..31,
// N = lane%16, halves j=0..15 map K offset j. Index:
//   Bp[ ((t*8 + kb)*32 + lane)*16 + j ] = Wv[(kb*32 + (lane>=16)*16 + j)*128 + t*16 + lane%16]
// ---------------------------------------------------------------------------
__global__ void gat_pack_B(const float* __restrict__ Wv, _Float16* __restrict__ Bp) {
    int i = blockIdx.x * blockDim.x + threadIdx.x;
    if (i >= BPACK_HALVES) return;
    int j  = i & 15;
    int l  = (i >> 4) & 31;
    int kb = (i >> 9) & 7;
    int t  = i >> 12;
    int K  = kb * 32 + (l >> 4) * 16 + j;
    int c  = t * 16 + (l & 15);
    Bp[i] = (_Float16)Wv[(size_t)K * CDIM + c];
}

// ---------------------------------------------------------------------------
// Kernel 1: h = x @ W_v via v_wmma_f32_16x16x32_f16, fused q/k epilogue.
// Block = 256 threads = 8 waves. Wave owns a DISTINCT 16-row M-tile and all
// 8 N-tiles (8 v8f accumulators) -> A read exactly once, 8 wmma per A load.
// B tiles are double-buffered one WMMA ahead to hide L2 latency.
// ---------------------------------------------------------------------------
__global__ __launch_bounds__(256) void gat_gemm_h(const float* __restrict__ x,
                                                  const _Float16* __restrict__ Bp,
                                                  const float* __restrict__ wq,
                                                  const float* __restrict__ wk,
                                                  float* __restrict__ h,
                                                  float* __restrict__ q,
                                                  float* __restrict__ kout,
                                                  int Mtiles) {
    const int lane = threadIdx.x & 31;
    const int wave = threadIdx.x >> 5;
    const int half = lane >> 4;             // lane group 0 / 1
    const int l16  = lane & 15;
    const int mt   = blockIdx.x * 8 + wave; // this wave's M-tile
    if (mt >= Mtiles) return;
    const int m0   = mt * 16;
    const int arow = m0 + l16;              // A: M striped across lanes

    v8f acc[8] = {{}, {}, {}, {}, {}, {}, {}, {}};

    const float* aprow = x + (size_t)arow * FDIM + half * 16;
    const v16h*  bptr  = (const v16h*)Bp + lane;   // + tile index * 32

    // prime the B pipeline: tile t=0 of kb=0
    v16h bcur = bptr[0 * 8 * 32 + 0 * 32];

    for (int kb = 0; kb < 8; ++kb) {
        // ---- A tile 16x32 f16: lane<16 -> K = kb*32+{0..15}; lane>=16 -> +16
        v16h a;
        const float* ap = aprow + kb * 32;
#pragma unroll
        for (int j = 0; j < 8; ++j) {
            a[j]     = (_Float16)ap[j];        // K = base + j
            a[j + 8] = (_Float16)ap[j + 8];    // K = base + 8 + j
        }
        // ---- 8 N-tiles, B double-buffered one wmma ahead
#pragma unroll
        for (int t = 0; t < 8; ++t) {
            v16h bnext;
            if (t < 7) {
                bnext = bptr[(size_t)((t + 1) * 8 + kb) * 32];       // next tile, same kb
            } else {
                int kbn = (kb < 7) ? (kb + 1) : 7;                    // tile 0 of next kb
                bnext = bptr[(size_t)(0 * 8 + kbn) * 32];
            }
            acc[t] = __builtin_amdgcn_wmma_f32_16x16x32_f16(
                /*neg_a=*/false, a, /*neg_b=*/false, bcur,
                /*c_mod=*/(short)0, acc[t], /*reuse_a=*/false, /*reuse_b=*/false);
            bcur = bnext;
        }
    }

    // ---- store h: VGPR r -> M = m0 + r + 8*half, N = t*16 + l16
    float* hp = h + (size_t)m0 * CDIM;
#pragma unroll
    for (int t = 0; t < 8; ++t) {
#pragma unroll
        for (int r = 0; r < 8; ++r) {
            hp[(size_t)(r + half * 8) * CDIM + t * 16 + l16] = acc[t][r];
        }
    }

    // ---- fused q/k: wave holds full 16x128 row block in registers
    float wqv[8], wkv[8];
#pragma unroll
    for (int t = 0; t < 8; ++t) {
        wqv[t] = wq[t * 16 + l16];
        wkv[t] = wk[t * 16 + l16];
    }
#pragma unroll
    for (int r = 0; r < 8; ++r) {
        float pq = 0.0f, pk = 0.0f;
#pragma unroll
        for (int t = 0; t < 8; ++t) {
            pq += acc[t][r] * wqv[t];
            pk += acc[t][r] * wkv[t];
        }
        // reduce across the 16 lanes of this half-group (wave32 shuffles)
#pragma unroll
        for (int mask = 1; mask < 16; mask <<= 1) {
            pq += __shfl_xor(pq, mask, 32);
            pk += __shfl_xor(pk, mask, 32);
        }
        if (l16 == 0) {
            q[m0 + half * 8 + r]    = pq;
            kout[m0 + half * 8 + r] = pk;
        }
    }
}

// ---------------------------------------------------------------------------
// Kernel 2: init out = bias, m = -inf, denom = 0
// ---------------------------------------------------------------------------
__global__ void gat_init(float* __restrict__ out, const float* __restrict__ bias,
                         float* __restrict__ m, float* __restrict__ denom, int N) {
    int i = blockIdx.x * blockDim.x + threadIdx.x;
    int total = N * CDIM;
    if (i < total) out[i] = bias[i & (CDIM - 1)];
    if (i < N) { m[i] = -__builtin_inff(); denom[i] = 0.0f; }
}

__device__ __forceinline__ void atomicMaxF32(float* addr, float val) {
    if (val >= 0.0f) atomicMax((int*)addr, __float_as_int(val));
    else             atomicMin((unsigned int*)addr, __float_as_uint(val));
}

// ---------------------------------------------------------------------------
// Kernel 3: edge logits: ee = leakyrelu(q[row]+k[col]); segment max into m
// ---------------------------------------------------------------------------
__global__ void gat_edge_logits(const long long* __restrict__ edges,
                                const float* __restrict__ q,
                                const float* __restrict__ k,
                                float* __restrict__ ee,
                                float* __restrict__ m, int E) {
    int e = blockIdx.x * blockDim.x + threadIdx.x;
    if (e >= E) return;
    int r = (int)edges[2 * (size_t)e];
    int c = (int)edges[2 * (size_t)e + 1];
    float v = q[r] + k[c];
    v = (v > 0.0f) ? v : LEAKY * v;
    ee[e] = v;
    atomicMaxF32(&m[r], v);
}

// ---------------------------------------------------------------------------
// Kernel 4: ee <- exp(ee - m[row]) in place; segment sum into denom
// ---------------------------------------------------------------------------
__global__ void gat_edge_exp(const long long* __restrict__ edges,
                             const float* __restrict__ m,
                             float* __restrict__ ee,
                             float* __restrict__ denom, int E) {
    int e = blockIdx.x * blockDim.x + threadIdx.x;
    if (e >= E) return;
    int r = (int)edges[2 * (size_t)e];
    float v = expf(ee[e] - m[r]);
    ee[e] = v;
    atomicAdd(&denom[r], v);
}

// ---------------------------------------------------------------------------
// Kernel 5: out[row] += (ee/denom[row]) * h[col]. One wave32 per edge, each
// lane owns 4 channels (float4 coalesced gather; h is L2-resident at 51 MB).
// ---------------------------------------------------------------------------
__global__ __launch_bounds__(256) void gat_edge_aggregate(
        const long long* __restrict__ edges,
        const float* __restrict__ ee,
        const float* __restrict__ denom,
        const float* __restrict__ h,
        float* __restrict__ out, int E) {
    const int lane = threadIdx.x & 31;
    const int e = blockIdx.x * 8 + (threadIdx.x >> 5);
    if (e >= E) return;
    int r = (int)edges[2 * (size_t)e];
    int c = (int)edges[2 * (size_t)e + 1];
    float d = denom[r];
    d = (d > 0.0f) ? d : 1.0f;
    float alpha = ee[e] / d;
    float4 hv = ((const float4*)(h + (size_t)c * CDIM))[lane];
    float* op = out + (size_t)r * CDIM + lane * 4;
    atomicAdd(op + 0, alpha * hv.x);
    atomicAdd(op + 1, alpha * hv.y);
    atomicAdd(op + 2, alpha * hv.z);
    atomicAdd(op + 3, alpha * hv.w);
}

// ---------------------------------------------------------------------------
extern "C" void kernel_launch(void* const* d_in, const int* in_sizes, int n_in,
                              void* d_out, int out_size, void* d_ws, size_t ws_size,
                              hipStream_t stream) {
    const float*      x     = (const float*)d_in[0];      // N*F
    const long long*  edges = (const long long*)d_in[1];  // E*2 (int64)
    const float*      Wv    = (const float*)d_in[2];      // F*C
    const float*      wq    = (const float*)d_in[3];      // C
    const float*      wk    = (const float*)d_in[4];      // C
    const float*      bias  = (const float*)d_in[5];      // C
    float*            out   = (float*)d_out;              // N*C

    const int N = in_sizes[0] / FDIM;
    const int E = in_sizes[1] / 2;
    const int Mtiles = N / 16;              // N = 100000 = 6250 * 16

    // workspace layout (256B aligned)
    char* ws = (char*)d_ws;
    size_t off = 0;
    auto alloc = [&](size_t bytes) {
        char* p = ws + off;
        off += (bytes + 255) & ~(size_t)255;
        return p;
    };
    _Float16* Bp  = (_Float16*)alloc((size_t)BPACK_HALVES * sizeof(_Float16)); // 64 KB
    float* h      = (float*)alloc((size_t)N * CDIM * sizeof(float));           // 51.2 MB
    float* q      = (float*)alloc((size_t)N * sizeof(float));
    float* k      = (float*)alloc((size_t)N * sizeof(float));
    float* m      = (float*)alloc((size_t)N * sizeof(float));
    float* denom  = (float*)alloc((size_t)N * sizeof(float));
    float* ee     = (float*)alloc((size_t)E * sizeof(float));                  // 12.8 MB
    (void)ws_size;

    // 0) swizzle W_v into WMMA B packing (f16)
    gat_pack_B<<<(BPACK_HALVES + 255) / 256, 256, 0, stream>>>(Wv, Bp);

    // 1) h = x @ W_v (WMMA) + fused q/k
    gat_gemm_h<<<(Mtiles + 7) / 8, 256, 0, stream>>>(x, Bp, wq, wk, h, q, k, Mtiles);

    // 2) init out=bias, m=-inf, denom=0
    {
        int total = N * CDIM;
        gat_init<<<(total + 255) / 256, 256, 0, stream>>>(out, bias, m, denom, N);
    }

    // 3) edge logits + segment max
    gat_edge_logits<<<(E + 255) / 256, 256, 0, stream>>>(edges, q, k, ee, m, E);

    // 4) exp + segment sum
    gat_edge_exp<<<(E + 255) / 256, 256, 0, stream>>>(edges, m, ee, denom, E);

    // 5) weighted scatter of h[col] into out[row]
    gat_edge_aggregate<<<(E + 7) / 8, 256, 0, stream>>>(edges, ee, denom, h, out, E);
}
